// ManualGRU_10806137717344
// MI455X (gfx1250) — matrix operand order repaired and checked by reading, not compiled, not run on hardware
//
#include <hip/hip_runtime.h>
#include <hip/hip_bf16.h>

// ---------------------------------------------------------------------------
// GRU on MI455X (gfx1250): bf16 WMMA (f32 accum) for both the big input
// projection GEMM and the sequential h @ W_h recurrence.
// Scan retiled so each block owns all 3 gate regions of its 16 hidden
// columns: hp stays in LDS, only ONE grid barrier per step. W_h fragments
// staged to LDS once via gfx1250 async-to-LDS (ASYNCcnt) ops.
// ---------------------------------------------------------------------------

typedef __attribute__((ext_vector_type(16))) __bf16 bf16x16; // A/B fragment (8 VGPRs)
typedef __attribute__((ext_vector_type(8)))  float  f32x8;   // C/D fragment (8 VGPRs)

#define B_    128
#define S_    1024
#define I_    256
#define H_    512
#define G3_   1536              // 3*H
#define M_    (B_ * S_)         // 131072 rows of x_proj
#define MT_   (M_ / 16)         // 8192 M tiles
#define NT_   (G3_ / 16)        // 96 N tiles
#define KT_IN (I_ / 32)         // 8 K tiles (input proj)
#define KT_H  (H_ / 32)         // 16 K tiles (recurrence)
#define SCAN_BLOCKS 32          // one block per 16 hidden columns (x3 gate regions)
#define SCAN_TPB    768         // 24 waves: (region r 0..2) x (m-tile 0..7)
#define HP_LD 49                // padded LDS row stride for hp (48 cols + 1)

// A-matrix 16x32 bf16 element->K mapping (ISA 7.12.2): lane half = lane>>4,
// element e (0..15): K = e<8 ? 8*half+e : 8*half+e+8
__device__ __forceinline__ int a_kloc(int e, int half) {
    return (e < 8) ? (8 * half + e) : (8 * half + e + 8);
}

__device__ __forceinline__ float sigmoidf_(float x) {
    return 1.0f / (1.0f + __expf(-x));
}

// --------------------------- grid-wide barrier ------------------------------
__device__ __forceinline__ void grid_barrier(unsigned* ctrs, unsigned nblocks) {
    __threadfence();
    __syncthreads();
    if (threadIdx.x == 0) {
        unsigned gen = __hip_atomic_load(&ctrs[1], __ATOMIC_RELAXED,
                                         __HIP_MEMORY_SCOPE_AGENT);
        unsigned arrived = __hip_atomic_fetch_add(&ctrs[0], 1u, __ATOMIC_ACQ_REL,
                                                  __HIP_MEMORY_SCOPE_AGENT);
        if (arrived == nblocks - 1u) {
            __hip_atomic_store(&ctrs[0], 0u, __ATOMIC_RELAXED,
                               __HIP_MEMORY_SCOPE_AGENT);
            __hip_atomic_fetch_add(&ctrs[1], 1u, __ATOMIC_RELEASE,
                                   __HIP_MEMORY_SCOPE_AGENT);
        } else {
            while (__hip_atomic_load(&ctrs[1], __ATOMIC_ACQUIRE,
                                     __HIP_MEMORY_SCOPE_AGENT) == gen) {
                __builtin_amdgcn_s_sleep(2);
            }
        }
        __threadfence();
    }
    __syncthreads();
}

// ------------------------------- init ---------------------------------------
__global__ void gru_init(unsigned* ctrs, float* hbuf, __bf16* hA) {
    int tid = blockIdx.x * blockDim.x + threadIdx.x;
    if (tid < 2) ctrs[tid] = 0u;
    for (int i = tid; i < B_ * H_; i += gridDim.x * blockDim.x) {
        hbuf[i] = 0.0f;            // h0 = 0
        hA[i]   = (__bf16)0.0f;    // packed A-fragments of h0 (65536 elems)
    }
}

// --------------------- pack weights into WMMA B layout -----------------------
// W is (K x N) row-major f32, N = 1536. Output: [n_tile][k_tile][lane][16] bf16.
// B layout (32x16): lanes 0-15 hold K=0..7,16..23, lanes 16-31 hold K=8..15,24..31.
__global__ void pack_wmma_B(const float* __restrict__ W, __bf16* __restrict__ out,
                            int k_tiles, int N) {
    int t = blockIdx.x * blockDim.x + threadIdx.x;
    int total = NT_ * k_tiles * 32;
    if (t >= total) return;
    int lane = t & 31;
    int kt   = (t >> 5) % k_tiles;
    int half = lane >> 4;
    int nt   = t / (32 * k_tiles);
    int n    = nt * 16 + (lane & 15);
    __bf16* dst = out + (size_t)t * 16;
#pragma unroll
    for (int e = 0; e < 16; ++e) {
        int k = kt * 32 + a_kloc(e, half);
        dst[e] = (__bf16)W[(size_t)k * N + n];
    }
}

// --------------------- pack activations into WMMA A layout -------------------
__global__ void pack_wmma_A(const float* __restrict__ X, __bf16* __restrict__ out,
                            int m_tiles, int k_tiles, int ldk) {
    int t = blockIdx.x * blockDim.x + threadIdx.x;
    int total = m_tiles * k_tiles * 32;
    if (t >= total) return;
    int lane = t & 31;
    int kt   = (t >> 5) % k_tiles;
    int half = lane >> 4;
    int mt   = t / (32 * k_tiles);
    int m    = mt * 16 + (lane & 15);
    __bf16* dst = out + (size_t)t * 16;
    const float* src = X + (size_t)m * ldk + kt * 32;
#pragma unroll
    for (int e = 0; e < 16; ++e)
        dst[e] = (__bf16)src[a_kloc(e, half)];
}

// --------------------- x_proj = X @ W_in + b_in (bf16 WMMA) ------------------
__global__ __launch_bounds__(256) void xproj_gemm(
        const __bf16* __restrict__ Ain, const __bf16* __restrict__ Winp,
        const float* __restrict__ bias, float* __restrict__ xproj) {
    int wave = (blockIdx.x * blockDim.x + threadIdx.x) >> 5;
    int lane = threadIdx.x & 31;
    int mt = wave / (NT_ / 4);       // 0..8191
    int ng = wave % (NT_ / 4);       // 0..23 (groups of 4 N tiles)
    if (mt >= MT_) return;           // uniform per wave -> EXEC stays full

    f32x8 acc[4] = {};
    const bf16x16* A  = (const bf16x16*)Ain;
    const bf16x16* Bp = (const bf16x16*)Winp;
#pragma unroll
    for (int kt = 0; kt < KT_IN; ++kt) {
        bf16x16 a = A[((size_t)mt * KT_IN + kt) * 32 + lane];
#pragma unroll
        for (int j = 0; j < 4; ++j) {
            bf16x16 b = Bp[((size_t)(ng * 4 + j) * KT_IN + kt) * 32 + lane];
            acc[j] = __builtin_amdgcn_wmma_f32_16x16x32_bf16(
                false, a, false, b, (short)0, acc[j], false, false);
        }
    }
    int row0 = mt * 16 + (lane >> 4) * 8;
#pragma unroll
    for (int j = 0; j < 4; ++j) {
        int n = (ng * 4 + j) * 16 + (lane & 15);
        float bv = bias[n];
#pragma unroll
        for (int v = 0; v < 8; ++v)
            xproj[(size_t)(row0 + v) * G3_ + n] = acc[j][v] + bv;
    }
}

// --------------------- persistent recurrent scan -----------------------------
// Block g (0..31) owns hidden columns [16g, 16g+16): it computes hp n-tiles
// {g, g+32, g+64} (all three gate regions) into LDS, so gates need only
// __syncthreads. One grid barrier per step (for globally shared packed hA).
__global__ __launch_bounds__(SCAN_TPB) void gru_scan(
        const __bf16* __restrict__ Whp, const float* __restrict__ xproj,
        const float* __restrict__ bias, float* __restrict__ hbuf,
        __bf16* __restrict__ hA, float* __restrict__ out_seq,
        float* __restrict__ out_last, unsigned* ctrs) {
    __shared__ __align__(32) __bf16 shW[3 * KT_H * 32 * 16]; // 48 KB B-fragments
    __shared__ float shHp[B_ * HP_LD];                       // 128 x 48 hp (padded)

    const int g    = blockIdx.x;         // 0..31
    const int wave = threadIdx.x >> 5;   // 0..23
    const int lane = threadIdx.x & 31;
    const int r    = wave / 8;           // gate region 0..2
    const int mt   = wave % 8;           // batch-row tile
    const float* b_h = bias + G3_;
    const bf16x16* A = (const bf16x16*)hA;

    // ---- stage this block's W_h B-fragments to LDS via async-to-LDS DMA ----
    {
        unsigned ldsbase = (unsigned)(uintptr_t)(void*)shW;
        const int bytesPerTile = KT_H * 32 * 32;             // 16 KB per n-tile
        for (int d = threadIdx.x * 16; d < 3 * bytesPerTile; d += SCAN_TPB * 16) {
            int rr = d / bytesPerTile;
            int o  = d % bytesPerTile;
            unsigned long long ga = (unsigned long long)(uintptr_t)
                ((const char*)Whp + (size_t)(32 * rr + g) * bytesPerTile + o);
            unsigned lds = ldsbase + (unsigned)d;
            asm volatile("global_load_async_to_lds_b128 %0, %1, off"
                         :: "v"(lds), "v"(ga) : "memory");
        }
        asm volatile("s_wait_asynccnt 0" ::: "memory");
    }
    __syncthreads();
    const bf16x16* shWB = (const bf16x16*)shW;

    for (int t = 0; t < S_; ++t) {
        // ---- phase 1: hp tile (rows 16mt.., gate-region r) = h @ W_h ----
        f32x8 acc = {};
#pragma unroll
        for (int kt = 0; kt < KT_H; ++kt) {
            bf16x16 a = A[((size_t)mt * KT_H + kt) * 32 + lane];
            bf16x16 b = shWB[(r * KT_H + kt) * 32 + lane];
            acc = __builtin_amdgcn_wmma_f32_16x16x32_bf16(
                false, a, false, b, (short)0, acc, false, false);
        }
        int lcol = r * 16 + (lane & 15);
        int row0 = mt * 16 + (lane >> 4) * 8;
#pragma unroll
        for (int v = 0; v < 8; ++v)
            shHp[(row0 + v) * HP_LD + lcol] = acc[v];

        __syncthreads();   // hp for our 16 columns fully in LDS

        // ---- phase 2: gates + write h_t + scatter bf16 h into A layout ----
        for (int idx = threadIdx.x; idx < B_ * 16; idx += SCAN_TPB) {
            int bb = idx >> 4;              // batch row 0..127
            int jl = idx & 15;              // local hidden col
            int j  = g * 16 + jl;           // global hidden col
            size_t xrow = ((size_t)bb * S_ + t) * G3_;
            float xr = xproj[xrow + j];
            float xz = xproj[xrow + H_ + j];
            float xn = xproj[xrow + 2 * H_ + j];
            float hr = shHp[bb * HP_LD + jl]      + b_h[j];
            float hz = shHp[bb * HP_LD + 16 + jl] + b_h[H_ + j];
            float hn = shHp[bb * HP_LD + 32 + jl] + b_h[2 * H_ + j];
            float rg = sigmoidf_(xr + hr);
            float zg = sigmoidf_(xz + hz);
            float ng = tanhf(xn + rg * hn);
            int hidx = bb * H_ + j;
            float hprev = hbuf[hidx];
            float hnew  = (1.0f - zg) * ng + zg * hprev;
            hbuf[hidx] = hnew;
            out_seq[((size_t)bb * S_ + t) * H_ + j] = hnew;
            if (t == S_ - 1) out_last[hidx] = hnew;
            // invert A-fragment mapping: (row bb, K=j) -> packed position
            int mt2 = bb >> 4, lm = bb & 15;
            int ktile = j >> 5, kloc = j & 31;
            int half = (kloc & 8) ? 1 : 0;
            int e = kloc - 8 * half - ((kloc >= 16) ? 8 : 0);
            hA[((size_t)(mt2 * KT_H + ktile) * 32 + (16 * half + lm)) * 16 + e] =
                (__bf16)hnew;
        }
        grid_barrier(ctrs, SCAN_BLOCKS);   // hA(t+1) globally visible
    }
}

// ---------------------------------------------------------------------------
extern "C" void kernel_launch(void* const* d_in, const int* in_sizes, int n_in,
                              void* d_out, int out_size, void* d_ws, size_t ws_size,
                              hipStream_t stream) {
    (void)in_sizes; (void)n_in; (void)out_size; (void)ws_size;
    const float* inputs = (const float*)d_in[0];   // (B,S,I)
    const float* W_in   = (const float*)d_in[1];   // (I,3H)
    const float* W_h    = (const float*)d_in[2];   // (H,3H)
    const float* bias   = (const float*)d_in[3];   // (6H,)
    float* out_seq  = (float*)d_out;                       // (B,S,H)
    float* out_last = out_seq + (size_t)B_ * S_ * H_;      // (B,H)

    // workspace layout (all offsets 256B aligned)
    char* ws = (char*)d_ws;
    unsigned* ctrs = (unsigned*)(ws + 0);              // 2 x u32 barrier state
    float*   hbuf  = (float*)  (ws + 256);             // 128x512 f32      (256 KB)
    __bf16*  hA    = (__bf16*) (ws + 262400);          // packed h A-frag  (128 KB)
    __bf16*  Whp   = (__bf16*) (ws + 393472);          // packed W_h B     (1.5 MB)
    __bf16*  Winp  = (__bf16*) (ws + 1966336);         // packed W_in B    (768 KB)
    __bf16*  Ainp  = (__bf16*) (ws + 3539200);         // packed inputs A  (64 MB)
    float*   xproj = (float*)  (ws + 70648064);        // 131072x1536 f32  (768 MB)

    gru_init<<<dim3(64), dim3(256), 0, stream>>>(ctrs, hbuf, hA);
    pack_wmma_B<<<dim3((NT_ * KT_IN * 32 + 255) / 256), dim3(256), 0, stream>>>(
        W_in, Winp, KT_IN, G3_);
    pack_wmma_B<<<dim3((NT_ * KT_H * 32 + 255) / 256), dim3(256), 0, stream>>>(
        W_h, Whp, KT_H, G3_);
    pack_wmma_A<<<dim3((MT_ * KT_IN * 32 + 255) / 256), dim3(256), 0, stream>>>(
        inputs, Ainp, MT_, KT_IN, I_);
    xproj_gemm<<<dim3(MT_ * (NT_ / 4) / 8), dim3(256), 0, stream>>>(
        Ainp, Winp, bias, xproj);
    gru_scan<<<dim3(SCAN_BLOCKS), dim3(SCAN_TPB), 0, stream>>>(
        Whp, xproj, bias, hbuf, hA, out_seq, out_last, ctrs);
}
